// GRUWrapper_70257075028382
// MI455X (gfx1250) — matrix operand order repaired and checked by reading, not compiled, not run on hardware
//
#include <hip/hip_runtime.h>
#include <math.h>

// ---------------------------------------------------------------------------
// GRU (ONNX linear_before_reset=0), SEQ=512, B=64, I=H=1024, fp32 in/out.
// bf16 WMMA (v_wmma_f32_16x16x32_bf16), f32 accumulate.
//   phase 1: gx = X * W_ih^T + b_ih        (one big GEMM, 32768x3072x1024)
//   phase 2: 512 sequential steps, 2 kernels/step (n-gate depends on r):
//       A: r = sig(gx_r + h Whr^T + b), z = sig(gx_z + h Whz^T + b)
//       B: n = tanh(gx_n + (r*h) Whn^T + b); h' = (1-z)n + z h -> Y[t]
// Strip loops are ping-pong register-pipelined (two explicit fragment sets):
// loads for k+64 issue while the k WMMAs execute, with no cur<-next copies.
// ---------------------------------------------------------------------------

#define SEQ 512
#define BATCH 64
#define IDIM 1024
#define HDIM 1024
#define G3H  (3 * HDIM)

typedef __attribute__((ext_vector_type(16))) __bf16 v16bf;
typedef __attribute__((ext_vector_type(8)))  __bf16 v8bf;
typedef __attribute__((ext_vector_type(8)))  float  v8f;

// Workspace layout (bytes, 256-aligned)
#define XBF_OFF   ((size_t)0)                         // SEQ*B*I bf16   = 67,108,864
#define WIH_OFF   ((size_t)67108864)                  // 3H*I bf16      =  6,291,456
#define WHH_OFF   ((size_t)73400320)                  // 3H*H bf16      =  6,291,456
#define GX_OFF    ((size_t)79691776)                  // SEQ*B*3H f32   = 402,653,184
#define HBF_OFF   ((size_t)482344960)                 // B*H bf16
#define RHBF_OFF  ((size_t)482476032)                 // B*H bf16
#define ZBUF_OFF  ((size_t)482607104)                 // B*H f32

__device__ __forceinline__ __bf16 to_bf16(float f) { return (__bf16)f; }

__global__ void f32_to_bf16_kernel(const float* __restrict__ src,
                                   __bf16* __restrict__ dst, int n) {
  int i = blockIdx.x * blockDim.x + threadIdx.x;
  int stride = gridDim.x * blockDim.x;
  for (; i < n; i += stride) dst[i] = to_bf16(src[i]);
}

// 16x32 bf16 A-fragment: lane<16 -> row=lane, K {0..15};
// lane>=16 -> row=lane-16, K {16..31}.  Two contiguous 16B loads.
__device__ __forceinline__ v16bf load_a_frag(const __bf16* __restrict__ arow,
                                             int k0, int half) {
  v8bf lo = *(const v8bf*)(arow + k0 + 16 * half);
  v8bf hi = *(const v8bf*)(arow + k0 + 16 * half + 8);
  return __builtin_shufflevector(lo, hi, 0, 1, 2, 3, 4, 5, 6, 7,
                                 8, 9, 10, 11, 12, 13, 14, 15);
}

#define WMMA_BF16(A, B, C) \
  __builtin_amdgcn_wmma_f32_16x16x32_bf16(false, (A), false, (B), (short)0, (C), false, false)

// ---------------------------------------------------------------------------
// Ping-pong pipelined 16(M) x 64(N) x K strip: 4 accumulators, A reused 4x.
// Two explicit fragment sets; loads for the other set issue before the WMMAs
// of the current set, so next-k loads stay in flight during XDL execution.
// Requires K % 64 == 0 (K = 1024 here).
// ---------------------------------------------------------------------------
__device__ __forceinline__ void wmma_strip_16x64(
    const __bf16* __restrict__ arow, int half,
    const __bf16* __restrict__ b0, const __bf16* __restrict__ b1,
    const __bf16* __restrict__ b2, const __bf16* __restrict__ b3,
    int K, v8f acc[4]) {
  // prologue: fragment set P at k=0
  v16bf aP  = load_a_frag(arow, 0, half);
  v16bf pB0 = *(const v16bf*)(b0);
  v16bf pB1 = *(const v16bf*)(b1);
  v16bf pB2 = *(const v16bf*)(b2);
  v16bf pB3 = *(const v16bf*)(b3);
  // fragment set Q at k=32
  v16bf aQ  = load_a_frag(arow, 32, half);
  v16bf qB0 = *(const v16bf*)(b0 + 32);
  v16bf qB1 = *(const v16bf*)(b1 + 32);
  v16bf qB2 = *(const v16bf*)(b2 + 32);
  v16bf qB3 = *(const v16bf*)(b3 + 32);

  for (int k0 = 64; k0 < K; k0 += 64) {
    // consume P(k0-64) while refilling P with k0
    acc[0] = WMMA_BF16(aP, pB0, acc[0]);
    acc[1] = WMMA_BF16(aP, pB1, acc[1]);
    acc[2] = WMMA_BF16(aP, pB2, acc[2]);
    acc[3] = WMMA_BF16(aP, pB3, acc[3]);
    aP  = load_a_frag(arow, k0, half);
    pB0 = *(const v16bf*)(b0 + k0);
    pB1 = *(const v16bf*)(b1 + k0);
    pB2 = *(const v16bf*)(b2 + k0);
    pB3 = *(const v16bf*)(b3 + k0);
    // consume Q(k0-32) while refilling Q with k0+32
    acc[0] = WMMA_BF16(aQ, qB0, acc[0]);
    acc[1] = WMMA_BF16(aQ, qB1, acc[1]);
    acc[2] = WMMA_BF16(aQ, qB2, acc[2]);
    acc[3] = WMMA_BF16(aQ, qB3, acc[3]);
    aQ  = load_a_frag(arow, k0 + 32, half);
    qB0 = *(const v16bf*)(b0 + k0 + 32);
    qB1 = *(const v16bf*)(b1 + k0 + 32);
    qB2 = *(const v16bf*)(b2 + k0 + 32);
    qB3 = *(const v16bf*)(b3 + k0 + 32);
  }
  // epilogue: drain both sets
  acc[0] = WMMA_BF16(aP, pB0, acc[0]);
  acc[1] = WMMA_BF16(aP, pB1, acc[1]);
  acc[2] = WMMA_BF16(aP, pB2, acc[2]);
  acc[3] = WMMA_BF16(aP, pB3, acc[3]);
  acc[0] = WMMA_BF16(aQ, qB0, acc[0]);
  acc[1] = WMMA_BF16(aQ, qB1, acc[1]);
  acc[2] = WMMA_BF16(aQ, qB2, acc[2]);
  acc[3] = WMMA_BF16(aQ, qB3, acc[3]);
}

// ---------------------------------------------------------------------------
// gx = Xbf(32768x1024) * Wihbf(3072x1024)^T + bias_ih   -> f32 (32768x3072)
// ---------------------------------------------------------------------------
__global__ __launch_bounds__(256) void gemm_gx_kernel(
    const __bf16* __restrict__ X, const __bf16* __restrict__ W,
    const float* __restrict__ bias, float* __restrict__ gx) {
  const int K = IDIM, N = G3H;
  const int NT = N / 64;                        // 48 n-strips
  int wave = (blockIdx.x * blockDim.x + threadIdx.x) >> 5;
  int lane = threadIdx.x & 31;
  int mt = wave / NT, nt = wave % NT;
  int m0 = mt * 16, n0 = nt * 64;
  int half = lane >> 4, r = lane & 15;

  v8f acc[4] = {v8f{}, v8f{}, v8f{}, v8f{}};
  const __bf16* arow = X + (size_t)(m0 + r) * K;
  wmma_strip_16x64(arow, half,
                   W + (size_t)(n0 + 0 + r) * K + 16 * half,
                   W + (size_t)(n0 + 16 + r) * K + 16 * half,
                   W + (size_t)(n0 + 32 + r) * K + 16 * half,
                   W + (size_t)(n0 + 48 + r) * K + 16 * half, K, acc);

  // C/D layout: VGPR v -> row m0 + v + 8*half, col n (lane%16)
#pragma unroll
  for (int j = 0; j < 4; ++j) {
    int n = n0 + j * 16 + r;
    float bj = bias[n];
#pragma unroll
    for (int v = 0; v < 8; ++v) {
      int m = m0 + v + 8 * half;
      gx[(size_t)m * N + n] = acc[j][v] + bj;
    }
  }
}

// ---------------------------------------------------------------------------
// Step phase A: gates r and z.  A = hbf (64x1024), B = Whh rows [0,2048).
// n < H  -> r gate: rhbf = bf16(sigmoid(pre) * h_prev)
// n >= H -> z gate: zbuf = sigmoid(pre)
// Grid is only 16 blocks -> occupancy is grid-limited, so allow the allocator
// a full register budget (min 1 block / WGP) to avoid accumulator shuffling.
// ---------------------------------------------------------------------------
__global__ __launch_bounds__(256, 1) void gru_gates_rz_kernel(
    const __bf16* __restrict__ hbf, const __bf16* __restrict__ Whh,
    const float* __restrict__ gx_t, const float* __restrict__ gx_next,
    const float* __restrict__ bias_hh, const float* __restrict__ hprev,
    float* __restrict__ zbuf, __bf16* __restrict__ rhbf) {
  const int K = HDIM;
  const int NT = 2048 / 64;                     // 32 n-strips
  int wave = (blockIdx.x * blockDim.x + threadIdx.x) >> 5;
  int lane = threadIdx.x & 31;
  int mt = wave / NT, nt = wave % NT;
  int m0 = mt * 16, n0 = nt * 64;
  int half = lane >> 4, r = lane & 15;

  v8f acc[4] = {v8f{}, v8f{}, v8f{}, v8f{}};
  const __bf16* arow = hbf + (size_t)(m0 + r) * K;
  wmma_strip_16x64(arow, half,
                   Whh + (size_t)(n0 + 0 + r) * K + 16 * half,
                   Whh + (size_t)(n0 + 16 + r) * K + 16 * half,
                   Whh + (size_t)(n0 + 32 + r) * K + 16 * half,
                   Whh + (size_t)(n0 + 48 + r) * K + 16 * half, K, acc);

#pragma unroll
  for (int j = 0; j < 4; ++j) {
    int n = n0 + j * 16 + r;                    // column in [0, 2048)
    float bh = bias_hh[n];
#pragma unroll
    for (int v = 0; v < 8; ++v) {
      int m = m0 + v + 8 * half;
      float pre = gx_t[(size_t)m * G3H + n] + acc[j][v] + bh;
      float s = 1.0f / (1.0f + __expf(-pre));
      if (n < HDIM) {
        rhbf[(size_t)m * HDIM + n] = to_bf16(s * hprev[(size_t)m * HDIM + n]);
      } else {
        zbuf[(size_t)m * HDIM + (n - HDIM)] = s;
      }
    }
  }

  // Warm L2 with the next timestep's gx slice for this wave's columns
  // (gx streams from HBM; lowers to global_prefetch_b8).
#pragma unroll
  for (int j = 0; j < 4; ++j) {
    int n = n0 + j * 16 + r;
#pragma unroll
    for (int v = 0; v < 8; v += 2) {
      int m = m0 + v + 8 * half;
      __builtin_prefetch(&gx_next[(size_t)m * G3H + n], 0, 1);
    }
  }
}

// ---------------------------------------------------------------------------
// Step phase B: gate n + state update.  A = rhbf, B = Whh rows [2H, 3H).
// h' = (1-z)*tanh(pre) + z*h_prev  -> Y[t] (f32) and hbf (bf16).
// ---------------------------------------------------------------------------
__global__ __launch_bounds__(256, 1) void gru_update_kernel(
    const __bf16* __restrict__ rhbf, const __bf16* __restrict__ WhhN,
    const float* __restrict__ gx_t, const float* __restrict__ gx_next,
    const float* __restrict__ bias_hh_n, const float* __restrict__ hprev,
    const float* __restrict__ zbuf, float* __restrict__ y_t,
    __bf16* __restrict__ hbf_out) {
  const int K = HDIM;
  const int NT = HDIM / 64;                     // 16 n-strips
  int wave = (blockIdx.x * blockDim.x + threadIdx.x) >> 5;
  int lane = threadIdx.x & 31;
  int mt = wave / NT, nt = wave % NT;
  int m0 = mt * 16, n0 = nt * 64;
  int half = lane >> 4, r = lane & 15;

  v8f acc[4] = {v8f{}, v8f{}, v8f{}, v8f{}};
  const __bf16* arow = rhbf + (size_t)(m0 + r) * K;
  wmma_strip_16x64(arow, half,
                   WhhN + (size_t)(n0 + 0 + r) * K + 16 * half,
                   WhhN + (size_t)(n0 + 16 + r) * K + 16 * half,
                   WhhN + (size_t)(n0 + 32 + r) * K + 16 * half,
                   WhhN + (size_t)(n0 + 48 + r) * K + 16 * half, K, acc);

#pragma unroll
  for (int j = 0; j < 4; ++j) {
    int n = n0 + j * 16 + r;
    float bh = bias_hh_n[n];
#pragma unroll
    for (int v = 0; v < 8; ++v) {
      int m = m0 + v + 8 * half;
      float pre = gx_t[(size_t)m * G3H + 2 * HDIM + n] + acc[j][v] + bh;
      float nv = tanhf(pre);
      float z = zbuf[(size_t)m * HDIM + n];
      float hp = hprev[(size_t)m * HDIM + n];
      float hn = (1.0f - z) * nv + z * hp;
      y_t[(size_t)m * HDIM + n] = hn;
      hbf_out[(size_t)m * HDIM + n] = to_bf16(hn);
    }
  }

  // Warm L2 with the next timestep's n-gate gx columns.
#pragma unroll
  for (int j = 0; j < 4; ++j) {
    int n = n0 + j * 16 + r;
#pragma unroll
    for (int v = 0; v < 8; v += 2) {
      int m = m0 + v + 8 * half;
      __builtin_prefetch(&gx_next[(size_t)m * G3H + 2 * HDIM + n], 0, 1);
    }
  }
}

__global__ void copy_f32_kernel(const float* __restrict__ src,
                                float* __restrict__ dst, int n) {
  int i = blockIdx.x * blockDim.x + threadIdx.x;
  int stride = gridDim.x * blockDim.x;
  for (; i < n; i += stride) dst[i] = src[i];
}

extern "C" void kernel_launch(void* const* d_in, const int* in_sizes, int n_in,
                              void* d_out, int out_size, void* d_ws, size_t ws_size,
                              hipStream_t stream) {
  const float* X   = (const float*)d_in[0];   // (SEQ, B, I)
  const float* h0  = (const float*)d_in[1];   // (1, B, H)
  const float* Wih = (const float*)d_in[2];   // (3H, I)
  const float* Whh = (const float*)d_in[3];   // (3H, H)
  const float* bih = (const float*)d_in[4];   // (3H,)
  const float* bhh = (const float*)d_in[5];   // (3H,)
  float* out = (float*)d_out;                 // Y (SEQ,1,B,H) then Y_h (1,B,H)

  char* ws = (char*)d_ws;
  __bf16* Xbf   = (__bf16*)(ws + XBF_OFF);
  __bf16* Wihbf = (__bf16*)(ws + WIH_OFF);
  __bf16* Whhbf = (__bf16*)(ws + WHH_OFF);
  float*  gx    = (float*)(ws + GX_OFF);
  __bf16* hbf   = (__bf16*)(ws + HBF_OFF);
  __bf16* rhbf  = (__bf16*)(ws + RHBF_OFF);
  float*  zbuf  = (float*)(ws + ZBUF_OFF);

  // --- precision conversion (bf16 inputs, f32 accumulate) ---
  f32_to_bf16_kernel<<<2048, 256, 0, stream>>>(X, Xbf, SEQ * BATCH * IDIM);
  f32_to_bf16_kernel<<<512, 256, 0, stream>>>(Wih, Wihbf, G3H * IDIM);
  f32_to_bf16_kernel<<<512, 256, 0, stream>>>(Whh, Whhbf, G3H * HDIM);
  f32_to_bf16_kernel<<<32, 256, 0, stream>>>(h0, hbf, BATCH * HDIM);

  // --- phase 1: gx = X W_ih^T + b_ih ---
  // waves = (32768/16) * (3072/64) = 98304 -> 12288 blocks of 8 waves
  gemm_gx_kernel<<<12288, 256, 0, stream>>>(Xbf, Wihbf, bih, gx);

  // --- phase 2: recurrent scan (Y slices double as the h-history) ---
  float* Y = out;
  for (int t = 0; t < SEQ; ++t) {
    const float* hprev = (t == 0) ? h0 : (Y + (size_t)(t - 1) * BATCH * HDIM);
    const float* gx_t = gx + (size_t)t * BATCH * G3H;
    const float* gx_n = gx + (size_t)((t + 1 < SEQ) ? t + 1 : t) * BATCH * G3H;
    // r,z: waves = 4 * 32 = 128 -> 16 blocks
    gru_gates_rz_kernel<<<16, 256, 0, stream>>>(hbf, Whhbf, gx_t, gx_n, bhh,
                                                hprev, zbuf, rhbf);
    // n + update: waves = 4 * 16 = 64 -> 8 blocks
    gru_update_kernel<<<8, 256, 0, stream>>>(
        rhbf, Whhbf + (size_t)2 * HDIM * HDIM, gx_t, gx_n, bhh + 2 * HDIM,
        hprev, zbuf, Y + (size_t)t * BATCH * HDIM, hbf);
  }

  // --- Y_h = h_{SEQ-1} ---
  copy_f32_kernel<<<64, 256, 0, stream>>>(
      Y + (size_t)(SEQ - 1) * BATCH * HDIM, out + (size_t)SEQ * BATCH * HDIM,
      BATCH * HDIM);
}
